// SCNN_46712064311873
// MI455X (gfx1250) — matrix operand order repaired and checked by reading, not compiled, not run on hardware
//
#include <hip/hip_runtime.h>
#include <hip/hip_bf16.h>

// ---------------------------------------------------------------------------
// Types for CDNA5 WMMA (wave32). bf16 A/B (v16bf = 8 VGPRs), f32 C/D (v8f).
// ---------------------------------------------------------------------------
typedef __bf16 bf16;
typedef __attribute__((ext_vector_type(16))) __bf16 v16bf;
typedef __attribute__((ext_vector_type(8)))  __bf16 v8bf;
typedef __attribute__((ext_vector_type(8)))  float  v8f;
typedef __attribute__((ext_vector_type(4)))  unsigned int u32x4;
typedef __attribute__((ext_vector_type(8)))  int   i32x8;
typedef __attribute__((ext_vector_type(4)))  int   i32x4;

#define B_N   8
#define H_N   36
#define W_N   100
#define C0    512
#define C1    1024
#define C2    128
#define HPAD  44     // 36 + 2*4
#define WPAD  108    // 100 + 2*4
#define HWP   (HPAD*WPAD)     // 4752
#define MPIX  (B_N*H_N*W_N)   // 28800
#define K1    (C0*9)          // 4608
#define EPSV  1e-5f
#define SEG_ELEMS (8*5*288*800)   // 9,216,000

// ---------------------------------------------------------------------------
// Fragment loaders (ISA 7.12.2 layouts), rows contiguous in K.
// A 16x32: lane row = lane&15 (M), hi = lane>>4; element i:
//   K = hi*8 + (i&7) + (i>=8?16:0)  -> 16B groups at K=hi*8 and K=16+hi*8.
// B 32x16: lane row = N, hi = lane>>4; element i: same K order as A.
// ---------------------------------------------------------------------------
__device__ inline v16bf frag_a(const bf16* rowp, int hi) {
  union { v16bf v; v8bf h[2]; } u;
  u.h[0] = *(const v8bf*)(rowp + hi * 8);
  u.h[1] = *(const v8bf*)(rowp + 16 + hi * 8);
  return u.v;
}
__device__ inline v16bf frag_b(const bf16* rowp, int hi) {
  union { v16bf v; v8bf h[2]; } u;
  u.h[0] = *(const v8bf*)(rowp + hi * 8);
  u.h[1] = *(const v8bf*)(rowp + 16 + hi * 8);
  return u.v;
}

// ---------------------------------------------------------------------------
// Packing kernels
// ---------------------------------------------------------------------------
__global__ __launch_bounds__(256)
void cvt_pad_x(const float* __restrict__ x, bf16* __restrict__ xp) {
  long idx = (long)blockIdx.x * 256 + threadIdx.x;   // 8*512*44*108
  int wp = (int)(idx % WPAD); long r = idx / WPAD;
  int hp = (int)(r % HPAD);  r /= HPAD;
  int ci = (int)(r % C0);    int b = (int)(r / C0);
  float v = 0.f;
  if (hp >= 4 && hp < 40 && wp >= 4 && wp < 104)
    v = x[(((long)b * C0 + ci) * H_N + (hp - 4)) * W_N + (wp - 4)];
  xp[idx] = (bf16)v;
}

// Tap-major K-permutation: d[n][t*512 + ci] = s[n][ci*9 + t]
__global__ __launch_bounds__(256)
void pack_w1a(const float* __restrict__ s, bf16* __restrict__ d) {
  long i = (long)blockIdx.x * 256 + threadIdx.x;     // 1024*4608
  int kk = (int)(i % K1); long n = i / K1;
  int t = kk >> 9, ci = kk & 511;                    // 512 = 2^9
  d[i] = (bf16)s[n * K1 + (long)ci * 9 + t];
}

__global__ __launch_bounds__(256)
void cvt_bf16(const float* __restrict__ s, bf16* __restrict__ d, long n) {
  long i = (long)blockIdx.x * 256 + threadIdx.x;
  if (i < n) d[i] = (bf16)s[i];
}

// ---------------------------------------------------------------------------
// Conv1 (dilated 3x3, 512->1024) as implicit GEMM with tap-major K order:
//   k' = tap*512 + ci, so each 32-slab has a fixed tap and the A-gather is a
//   constant-stride walk (no divides in the hot loop).
// Block tile 64(M) x 128(N), K-step 32; 8 waves (2x4); wave = 2x2 WMMA tiles.
// Epilogue: BN1 + ReLU -> h1 bf16 [m][1024].
// ---------------------------------------------------------------------------
__global__ __launch_bounds__(256)
void conv1_gemm(const bf16* __restrict__ xpad, const bf16* __restrict__ wperm,
                const float* __restrict__ g1, const float* __restrict__ b1,
                const float* __restrict__ m1, const float* __restrict__ v1,
                bf16* __restrict__ h1) {
  __shared__ bf16 At[64 * 32];
  __shared__ bf16 Bt[128 * 32];
  const int m0 = blockIdx.x * 64;
  const int n0 = blockIdx.y * 128;
  const int tid = threadIdx.x;
  const int lane = tid & 31, wid = tid >> 5;
  const int wm = wid >> 2, wn = wid & 3;     // 2 x 4 wave grid
  const int row = lane & 15, hi = lane >> 4;

  // A-gather mapping (fixed per thread): m_l = tid&63, k-group = tid>>6
  const int a_ml = tid & 63, a_kg = tid >> 6;
  const int m = m0 + a_ml;
  const int pb = m / (H_N * W_N);
  const int pr = m - pb * (H_N * W_N);
  const int ph = pr / W_N, pw = pr - (pr / W_N) * W_N;
  const bf16* xb = xpad + (size_t)pb * C0 * HWP;

  v8f acc[2][2] = {};

  for (int t = 0; t < 9; ++t) {
    const int kh = t / 3, kw = t - (t / 3) * 3;
    // per-thread gather base for this tap (includes this thread's k-group)
    const bf16* abase = xb + (ph + kh * 4) * WPAD + (pw + kw * 4)
                           + (size_t)(a_kg * 8) * HWP;
    for (int cb = 0; cb < C0; cb += 32) {
      const int k0 = t * C0 + cb;            // linear permuted-K base
      // ---- stage A: 8 strided elements, ci = cb + a_kg*8 + j ----
#pragma unroll
      for (int j = 0; j < 8; ++j)
        At[a_ml * 32 + a_kg * 8 + j] = abase[(size_t)(cb + j) * HWP];
      // ---- stage B: contiguous rows of permuted w1a ----
#pragma unroll
      for (int s = tid; s < 512; s += 256) {
        int n = s >> 2, gk = s & 3;
        *(v8bf*)&Bt[n * 32 + gk * 8] =
            *(const v8bf*)&wperm[(size_t)(n0 + n) * K1 + k0 + gk * 8];
        if (k0 + 32 < K1)
          __builtin_prefetch(&wperm[(size_t)(n0 + n) * K1 + k0 + 32 + gk * 8], 0, 1);
      }
      __syncthreads();
      // ---- MMA ----
      v16bf af[2], bfr[2];
#pragma unroll
      for (int i = 0; i < 2; ++i) af[i]  = frag_a(&At[(wm * 32 + i * 16 + row) * 32], hi);
#pragma unroll
      for (int i = 0; i < 2; ++i) bfr[i] = frag_b(&Bt[(wn * 32 + i * 16 + row) * 32], hi);
#pragma unroll
      for (int i = 0; i < 2; ++i)
#pragma unroll
        for (int j = 0; j < 2; ++j)
          acc[i][j] = __builtin_amdgcn_wmma_f32_16x16x32_bf16(
              false, af[i], false, bfr[j], (short)0, acc[i][j], false, false);
      __syncthreads();
    }
  }
  // ---- epilogue: BN1 + ReLU, store bf16 h1[m][n] ----
#pragma unroll
  for (int i = 0; i < 2; ++i) {
    int mbase = m0 + wm * 32 + i * 16 + hi * 8;
#pragma unroll
    for (int j = 0; j < 2; ++j) {
      int n = n0 + wn * 32 + j * 16 + row;
      float sc = g1[n] * rsqrtf(v1[n] + EPSV);
      float bi = b1[n] - m1[n] * sc;
#pragma unroll
      for (int e = 0; e < 8; ++e) {
        float val = fmaxf(acc[i][j][e] * sc + bi, 0.f);
        h1[(size_t)(mbase + e) * C1 + n] = (bf16)val;
      }
    }
  }
}

// ---------------------------------------------------------------------------
// Conv1b (1x1, 1024->128): the whole bf16 w1b (256 KB) is DMA'd into LDS once
// by the Tensor Data Mover, then the K-loop runs barrier-free:
//   A fragments load directly from global h1 (rows contiguous in K),
//   B fragments read from resident LDS.
// D# per ISA 8.3/8.4: 2D tensor = tile = 1024 x 128, data_size=2B, stride 1024,
// lds_addr = 0 (Bt is the only LDS object).  Epilogue: BN2+ReLU -> h2 NCHW f32.
// ---------------------------------------------------------------------------
__global__ __launch_bounds__(256)
void conv1b_gemm(const bf16* __restrict__ h1, const bf16* __restrict__ wbb,
                 const float* __restrict__ g2, const float* __restrict__ b2,
                 const float* __restrict__ m2, const float* __restrict__ v2,
                 float* __restrict__ h2) {
  __shared__ bf16 Bt[C2 * C1];           // 256 KB resident B
  const int tid = threadIdx.x;
  const int lane = tid & 31, wid = tid >> 5;
  const int wm = wid >> 2, wn = wid & 3;
  const int row = lane & 15, hi = lane >> 4;
  const int m0 = blockIdx.x * 64;

  if (tid < 32) {                        // wave 0 issues the TDM load
    unsigned long long ga = (unsigned long long)(uintptr_t)wbb;
    u32x4 g0;
    g0[0] = 1u;                                          // count=1, user desc
    g0[1] = 0u;                                          // lds_addr = 0
    g0[2] = (unsigned)(ga & 0xFFFFFFFFull);              // global_addr lo
    g0[3] = (unsigned)((ga >> 32) & 0x1FFFFFFull) | (2u << 30);  // hi | type=2
    i32x8 g1;
    g1[0] = (int)(1u << 16);                             // data_size=2B, mask=0
    g1[1] = (int)((1024u & 0xFFFFu) << 16);              // tensor_dim0 lo16
    g1[2] = (int)(((1024u >> 16) & 0xFFFFu) | ((128u & 0xFFFFu) << 16)); // d0 hi|d1 lo
    g1[3] = (int)(((128u >> 16) & 0xFFFFu) | ((1024u & 0xFFFFu) << 16)); // d1 hi|tile0
    g1[4] = (int)(128u);                                 // tile_dim1 | tile_dim2=0
    g1[5] = (int)(1024u);                                // dim0_stride lo32
    g1[6] = 0;                                           // stride hi | d1stride lo
    g1[7] = 0;
    i32x4 gz4 = {0, 0, 0, 0};
    i32x8 gz8 = {0, 0, 0, 0, 0, 0, 0, 0};
    __builtin_amdgcn_tensor_load_to_lds(g0, g1, gz4, gz4, gz8, 0);
    __builtin_amdgcn_s_wait_tensorcnt(0);
  }
  __syncthreads();

  v8f acc[2][2] = {};
  for (int k0 = 0; k0 < C1; k0 += 32) {
    v16bf af[2], bfr[2];
#pragma unroll
    for (int i = 0; i < 2; ++i) {        // A straight from global (contiguous)
      const bf16* ap = h1 + (size_t)(m0 + wm * 32 + i * 16 + row) * C1 + k0;
      union { v16bf v; v8bf h[2]; } u;
      u.h[0] = *(const v8bf*)(ap + hi * 8);
      u.h[1] = *(const v8bf*)(ap + 16 + hi * 8);
      af[i] = u.v;
    }
#pragma unroll
    for (int i = 0; i < 2; ++i)
      bfr[i] = frag_b(&Bt[(size_t)(wn * 32 + i * 16 + row) * C1 + k0], hi);
#pragma unroll
    for (int i = 0; i < 2; ++i)
#pragma unroll
      for (int j = 0; j < 2; ++j)
        acc[i][j] = __builtin_amdgcn_wmma_f32_16x16x32_bf16(
            false, af[i], false, bfr[j], (short)0, acc[i][j], false, false);
  }
#pragma unroll
  for (int i = 0; i < 2; ++i) {
    int mbase = m0 + wm * 32 + i * 16 + hi * 8;
#pragma unroll
    for (int j = 0; j < 2; ++j) {
      int n = wn * 32 + j * 16 + row;    // 0..127
      float sc = g2[n] * rsqrtf(v2[n] + EPSV);
      float bi = b2[n] - m2[n] * sc;
#pragma unroll
      for (int e = 0; e < 8; ++e) {
        int mm = mbase + e;
        int pb = mm / (H_N * W_N);
        int pr = mm - pb * (H_N * W_N);
        int ph = pr / W_N, pw = pr - ph * W_N;
        float val = fmaxf(acc[i][j][e] * sc + bi, 0.f);
        h2[(((size_t)pb * C2 + n) * H_N + ph) * W_N + pw] = val;
      }
    }
  }
}

// ---------------------------------------------------------------------------
// Message passing: sequential directional conv1d steps (f32, LDS-staged).
//   h[cur] += relu(conv1d(h[prv], wt)), kernel 9, pad 4.
// ---------------------------------------------------------------------------
__global__ __launch_bounds__(256)
void mp_v_step(float* __restrict__ h, const float* __restrict__ wt, int prv, int cur) {
  __shared__ float P[C2][WPAD];           // prev row, W padded by 4
  const int b = blockIdx.x, gco = blockIdx.y, tid = threadIdx.x;
  float* hb = h + (size_t)b * C2 * H_N * W_N;
  for (int i = tid; i < C2 * 8; i += 256) {      // zero pad columns
    int ci = i >> 3, j = i & 7;
    P[ci][j < 4 ? j : 100 + j] = 0.f;
  }
  for (int i = tid; i < C2 * W_N; i += 256) {
    int ci = i / W_N, w = i - (i / W_N) * W_N;
    P[ci][w + 4] = hb[((size_t)ci * H_N + prv) * W_N + w];
  }
  __syncthreads();
  for (int o = tid; o < 32 * W_N; o += 256) {
    int col = o / W_N, w = o - col * W_N;
    int co = gco * 32 + col;
    const float* wr = wt + (size_t)co * C2 * 9;
    float acc = 0.f;
    for (int ci = 0; ci < C2; ++ci) {
      const float* pr = &P[ci][w];
      const float* ww = wr + ci * 9;
#pragma unroll
      for (int t = 0; t < 9; ++t) acc += pr[t] * ww[t];
    }
    float* dst = &hb[((size_t)co * H_N + cur) * W_N + w];
    *dst = *dst + fmaxf(acc, 0.f);
  }
}

__global__ __launch_bounds__(256)
void mp_h_step(float* __restrict__ h, const float* __restrict__ wt, int prv, int cur) {
  __shared__ float P[C2][HPAD];           // prev column, H padded by 4
  const int b = blockIdx.x, gco = blockIdx.y, tid = threadIdx.x;
  float* hb = h + (size_t)b * C2 * H_N * W_N;
  for (int i = tid; i < C2 * 8; i += 256) {
    int ci = i >> 3, j = i & 7;
    P[ci][j < 4 ? j : 36 + j] = 0.f;
  }
  for (int i = tid; i < C2 * H_N; i += 256) {
    int ci = i / H_N, r = i - (i / H_N) * H_N;
    P[ci][r + 4] = hb[((size_t)ci * H_N + r) * W_N + prv];
  }
  __syncthreads();
  for (int o = tid; o < 32 * H_N; o += 256) {
    int col = o / H_N, r = o - col * H_N;
    int co = gco * 32 + col;
    const float* wr = wt + (size_t)co * C2 * 9;
    float acc = 0.f;
    for (int ci = 0; ci < C2; ++ci) {
      const float* pr = &P[ci][r];
      const float* ww = wr + ci * 9;
#pragma unroll
      for (int t = 0; t < 9; ++t) acc += pr[t] * ww[t];
    }
    float* dst = &hb[((size_t)co * H_N + r) * W_N + cur];
    *dst = *dst + fmaxf(acc, 0.f);
  }
}

// ---------------------------------------------------------------------------
// 1x1 conv to 5 channels + bias
// ---------------------------------------------------------------------------
__global__ __launch_bounds__(256)
void conv2_k(const float* __restrict__ h2, const float* __restrict__ w2,
             const float* __restrict__ b2, float* __restrict__ h3) {
  int idx = blockIdx.x * 256 + threadIdx.x;          // 8*5*36*100
  if (idx >= B_N * 5 * H_N * W_N) return;
  int w = idx % W_N; int r = idx / W_N;
  int hh = r % H_N;  r /= H_N;
  int o = r % 5;     int b = r / 5;
  const float* hb = h2 + ((size_t)b * C2 * H_N + hh) * W_N + w;
  const float* wr = w2 + o * C2;
  float acc = b2[o];
  for (int ci = 0; ci < C2; ++ci) acc += hb[(size_t)ci * H_N * W_N] * wr[ci];
  h3[idx] = acc;
}

// ---------------------------------------------------------------------------
// Bilinear upsample x8, align-corners -> seg_pred (d_out[0 .. 9,216,000))
// ---------------------------------------------------------------------------
__global__ __launch_bounds__(256)
void upsample_k(const float* __restrict__ h3, float* __restrict__ out) {
  long idx = (long)blockIdx.x * 256 + threadIdx.x;   // 9,216,000
  if (idx >= (long)SEG_ELEMS) return;
  int wo = (int)(idx % 800); long r = idx / 800;
  int ho = (int)(r % 288);   r /= 288;
  int c = (int)(r % 5);      int b = (int)(r / 5);
  float py = ho * (35.f / 287.f);
  int i0 = (int)py; if (i0 > 34) i0 = 34;
  float fy = py - (float)i0;
  float px = wo * (99.f / 799.f);
  int j0 = (int)px; if (j0 > 98) j0 = 98;
  float fx = px - (float)j0;
  const float* p = h3 + (((size_t)b * 5 + c) * H_N + i0) * W_N + j0;
  float v00 = p[0], v01 = p[1], v10 = p[W_N], v11 = p[W_N + 1];
  float vt = v00 + (v01 - v00) * fx;
  float vb = v10 + (v11 - v10) * fx;
  out[idx] = vt + (vb - vt) * fy;
}

// ---------------------------------------------------------------------------
// Softmax over 5 channels + 2x2 mean pool -> p [8][4500] (c*900 + ph*50 + pw)
// ---------------------------------------------------------------------------
__global__ __launch_bounds__(256)
void smpool_k(const float* __restrict__ h3, float* __restrict__ p) {
  int idx = blockIdx.x * 256 + threadIdx.x;          // 8*18*50
  if (idx >= B_N * 18 * 50) return;
  int pw = idx % 50; int r = idx / 50;
  int ph = r % 18;   int b = r / 18;
  float s[5] = {0.f, 0.f, 0.f, 0.f, 0.f};
  for (int dy = 0; dy < 2; ++dy)
    for (int dx = 0; dx < 2; ++dx) {
      int hh = ph * 2 + dy, ww = pw * 2 + dx;
      float v[5], mx = -1e30f;
      for (int c = 0; c < 5; ++c) {
        v[c] = h3[(((size_t)b * 5 + c) * H_N + hh) * W_N + ww];
        mx = fmaxf(mx, v[c]);
      }
      float sum = 0.f;
      for (int c = 0; c < 5; ++c) { v[c] = __expf(v[c] - mx); sum += v[c]; }
      float inv = 1.f / sum;
      for (int c = 0; c < 5; ++c) s[c] += v[c] * inv;
    }
  for (int c = 0; c < 5; ++c)
    p[(size_t)b * 4500 + c * 900 + ph * 50 + pw] = s[c] * 0.25f;
}

// ---------------------------------------------------------------------------
// FC1(4500->128)+ReLU, FC2(128->4)+sigmoid -> exist_pred (d_out tail)
// ---------------------------------------------------------------------------
__global__ __launch_bounds__(128)
void fc_k(const float* __restrict__ p, const float* __restrict__ w1,
          const float* __restrict__ b1, const float* __restrict__ w2,
          const float* __restrict__ b2, float* __restrict__ out_exist) {
  __shared__ float f[128];
  int b = blockIdx.x, j = threadIdx.x;
  const float* pb = p + (size_t)b * 4500;
  const float* wr = w1 + (size_t)j * 4500;
  float acc = b1[j];
  for (int i = 0; i < 4500; ++i) acc += pb[i] * wr[i];
  f[j] = fmaxf(acc, 0.f);
  __syncthreads();
  if (j < 4) {
    float a2 = b2[j];
    for (int k = 0; k < 128; ++k) a2 += f[k] * w2[j * 128 + k];
    out_exist[b * 4 + j] = 1.f / (1.f + __expf(-a2));
  }
}

// ---------------------------------------------------------------------------
// Host launcher
// ---------------------------------------------------------------------------
static inline size_t al256(size_t x) { return (x + 255) & ~(size_t)255; }

extern "C" void kernel_launch(void* const* d_in, const int* in_sizes, int n_in,
                              void* d_out, int out_size, void* d_ws, size_t ws_size,
                              hipStream_t stream) {
  const float* x     = (const float*)d_in[0];
  const float* w1a   = (const float*)d_in[1];
  const float* bn1_g = (const float*)d_in[2];
  const float* bn1_b = (const float*)d_in[3];
  const float* bn1_m = (const float*)d_in[4];
  const float* bn1_v = (const float*)d_in[5];
  const float* w1b   = (const float*)d_in[6];
  const float* bn2_g = (const float*)d_in[7];
  const float* bn2_b = (const float*)d_in[8];
  const float* bn2_m = (const float*)d_in[9];
  const float* bn2_v = (const float*)d_in[10];
  const float* w_ud  = (const float*)d_in[11];
  const float* w_du  = (const float*)d_in[12];
  const float* w_lr  = (const float*)d_in[13];
  const float* w_rl  = (const float*)d_in[14];
  const float* w2    = (const float*)d_in[15];
  const float* b2    = (const float*)d_in[16];
  const float* fc1_w = (const float*)d_in[17];
  const float* fc1_b = (const float*)d_in[18];
  const float* fc2_w = (const float*)d_in[19];
  const float* fc2_b = (const float*)d_in[20];
  float* out = (float*)d_out;

  // workspace carve-out
  char* ws = (char*)d_ws;
  size_t off = 0;
  const long n_xpad = (long)B_N * C0 * HPAD * WPAD;       // 19,464,192
  bf16* xpad = (bf16*)(ws + off); off = al256(off + (size_t)n_xpad * 2);
  const long n_w1a = (long)C1 * K1;                       // 4,718,592
  bf16* w1aperm = (bf16*)(ws + off); off = al256(off + (size_t)n_w1a * 2);
  bf16* h1 = (bf16*)(ws + off); off = al256(off + (size_t)MPIX * C1 * 2);
  const long n_w1b = (long)C2 * C1;                       // 131,072
  bf16* w1bbf = (bf16*)(ws + off); off = al256(off + (size_t)n_w1b * 2);
  float* h2 = (float*)(ws + off); off = al256(off + (size_t)B_N * C2 * H_N * W_N * 4);
  float* h3 = (float*)(ws + off); off = al256(off + (size_t)B_N * 5 * H_N * W_N * 4);
  float* pp = (float*)(ws + off); off = al256(off + (size_t)B_N * 4500 * 4);

  // 1) pack inputs to bf16 (w1a in tap-major K order)
  cvt_pad_x<<<(int)(n_xpad / 256), 256, 0, stream>>>(x, xpad);
  pack_w1a<<<(int)((n_w1a + 255) / 256), 256, 0, stream>>>(w1a, w1aperm);
  cvt_bf16<<<(int)((n_w1b + 255) / 256), 256, 0, stream>>>(w1b, w1bbf, n_w1b);

  // 2) dilated 3x3 conv (implicit GEMM, WMMA bf16) + BN1 + ReLU
  conv1_gemm<<<dim3(MPIX / 64, C1 / 128), 256, 0, stream>>>(
      xpad, w1aperm, bn1_g, bn1_b, bn1_m, bn1_v, h1);

  // 3) 1x1 conv 1024->128 (WMMA bf16, TDM-resident B) + BN2 + ReLU -> h2
  conv1b_gemm<<<dim3(MPIX / 64, 1), 256, 0, stream>>>(
      h1, w1bbf, bn2_g, bn2_b, bn2_m, bn2_v, h2);

  // 4) directional message passing (sequential)
  for (int i = 1; i < H_N; ++i)
    mp_v_step<<<dim3(B_N, 4), 256, 0, stream>>>(h2, w_ud, i - 1, i);
  for (int i = H_N - 2; i >= 0; --i)
    mp_v_step<<<dim3(B_N, 4), 256, 0, stream>>>(h2, w_du, i + 1, i);
  for (int j = 1; j < W_N; ++j)
    mp_h_step<<<dim3(B_N, 4), 256, 0, stream>>>(h2, w_lr, j - 1, j);
  for (int j = W_N - 2; j >= 0; --j)
    mp_h_step<<<dim3(B_N, 4), 256, 0, stream>>>(h2, w_rl, j + 1, j);

  // 5) 1x1 conv to 5 ch + bias
  conv2_k<<<(B_N * 5 * H_N * W_N + 255) / 256, 256, 0, stream>>>(h2, w2, b2, h3);

  // 6) bilinear x8 upsample -> seg_pred
  upsample_k<<<SEG_ELEMS / 256, 256, 0, stream>>>(h3, out);

  // 7) softmax + 2x2 pool -> p ; FCs -> exist_pred
  smpool_k<<<(B_N * 18 * 50 + 255) / 256, 256, 0, stream>>>(h3, pp);
  fc_k<<<B_N, 128, 0, stream>>>(pp, fc1_w, fc1_b, fc2_w, fc2_b, out + SEG_ELEMS);

  (void)in_sizes; (void)n_in; (void)out_size; (void)ws_size;
}
